// Convolution_74302934221226
// MI455X (gfx1250) — matrix operand order repaired and checked by reading, not compiled, not run on hardware
//
#include <hip/hip_runtime.h>
#include <hip/hip_bf16.h>

// ---------------- problem constants (match reference) ----------------
#define NB    8
#define NCIN  32
#define NH    64
#define NW    64
#define NCOUT 64
#define NK    8
#define NGADD 4
#define NRADD 4
#define NVS   12          // 4 near + 4 global + 4 local
#define NHID  128
#define NADIN 34          // CIN + 2 coords
#define NPIX  (NB*NH*NW)  // 32768
#define KDIM  (NK*NCIN)   // 256
#define W1PAD 36          // padded row length for transposed W1 in LDS

typedef __attribute__((ext_vector_type(16))) _Float16 v16h;
typedef __attribute__((ext_vector_type(8)))  _Float16 v8h;
typedef __attribute__((ext_vector_type(4)))  _Float16 v4h;
typedef __attribute__((ext_vector_type(8)))  float    v8f;
typedef __attribute__((ext_vector_type(4)))  float    v4f;

__device__ __forceinline__ float softplus_f(float x) {
    return (x > 20.f) ? x : log1pf(expf(x));
}

// ---------------------------------------------------------------------
// Kernel 0: transpose x (B,CIN,H,W) -> xT (B*H*W, CIN) so every gather
// point is a 128B-contiguous row (8x global_load_b128 instead of 32
// strided scalar loads). Reads are fully coalesced across the block.
// ---------------------------------------------------------------------
__global__ __launch_bounds__(256)
void transpose_x_kernel(const float* __restrict__ x, float* __restrict__ xT)
{
    int gid = blockIdx.x * 256 + threadIdx.x;   // flat (b,h,w)
    int b = gid >> 12;
    int p = gid & 4095;
    const float* xb = x + (((size_t)b * NCIN) << 12) + p;
    float* dst = xT + (size_t)gid * NCIN;
    #pragma unroll
    for (int c = 0; c < NCIN; ++c) dst[c] = xb[(size_t)c << 12];
}

// ---------------------------------------------------------------------
// Stage 1: per-pixel MLP -> gaussian mixture sampling -> weighted gather.
// One thread per pixel. FUSED=false: read pixel-major xT, write f16 feats
// rows to workspace. FUSED=true: no workspace; gather channel-major x and
// apply the final 256x64 GEMM in VALU (fallback path).
// ---------------------------------------------------------------------
template<bool FUSED>
__global__ __launch_bounds__(256)
void stage1_kernel(const float* __restrict__ x,
                   const float* __restrict__ xT,
                   const float* __restrict__ W1, const float* __restrict__ b1,
                   const float* __restrict__ W2, const float* __restrict__ b2,
                   const int*   __restrict__ gidx, const int* __restrict__ loff,
                   unsigned short* __restrict__ feats_raw,
                   const float* __restrict__ Wu, const float* __restrict__ bu,
                   float* __restrict__ out)
{
    // W1 stored TRANSPOSED + padded: sW1T[i*36 + c] = W1[c*128 + i]
    // -> inner dot product over c is contiguous, vectorizes to ds_load_b128.
    __shared__ __align__(16) float sW1T[NHID * W1PAD];   // 18432 B
    __shared__ __align__(16) float sW2[NHID * NK * 3];   // 12288 B
    __shared__ float sb1[NHID];
    __shared__ float sb2[NK * 3];

    int t = threadIdx.x;
    for (int idx = t; idx < NHID * NADIN; idx += 256) {
        int i = idx / NADIN;
        int c = idx - i * NADIN;
        sW1T[i * W1PAD + c] = W1[c * NHID + i];
    }
    for (int i = t; i < NHID * NK * 3; i += 256) sW2[i] = W2[i];
    if (t < NHID)  sb1[t] = b1[t];
    if (t < NK*3)  sb2[t] = b2[t];
    __syncthreads();

    int gid = blockIdx.x * 256 + t;        // flat (b,h,w)
    int b = gid >> 12;
    int p = gid & 4095;
    int h = p >> 6;
    int w = p & 63;

    // ---- input vector: 32 channels + 2 normalized coords ----
    const float* xb = x + (((size_t)b * NCIN) << 12);   // channel-major base
    float inp[NADIN];
    if constexpr (FUSED) {
        #pragma unroll
        for (int c = 0; c < NCIN; ++c) inp[c] = xb[(c << 12) + p];
    } else {
        const float* xrow = xT + (size_t)gid * NCIN;    // contiguous 128B
        #pragma unroll
        for (int c = 0; c < NCIN; ++c) inp[c] = xrow[c];
    }
    inp[32] = (float)h * (1.f/63.f);
    inp[33] = (float)w * (1.f/63.f);

    // ---- MLP: relu(inp@W1+b1)@W2+b2, hidden streamed ----
    float par[NK*3];
    #pragma unroll
    for (int j = 0; j < NK*3; ++j) par[j] = sb2[j];
    for (int i = 0; i < NHID; ++i) {
        float hv = sb1[i];
        const float* wr = &sW1T[i * W1PAD];
        #pragma unroll
        for (int c = 0; c < NADIN; ++c) hv += inp[c] * wr[c];
        hv = fmaxf(hv, 0.f);
        const float* w2r = &sW2[i * (NK*3)];
        #pragma unroll
        for (int j = 0; j < NK*3; ++j) par[j] += hv * w2r[j];
    }

    float outacc[NCOUT];
    if constexpr (FUSED) {
        #pragma unroll
        for (int co = 0; co < NCOUT; ++co) outacc[co] = bu[co];
    }

    const int* gbase = gidx + (size_t)gid * (NK*NGADD*2);
    const int* lbase = loff + (size_t)gid * (NK*NRADD*2);

    for (int k = 0; k < NK; ++k) {
        // means = mod((h,w) + raw, 63); sigma = (softplus(s+2)+0.05)*63*0.05
        float my = fmodf((float)h + par[2*k],   63.f); if (my < 0.f) my += 63.f;
        float mx = fmodf((float)w + par[2*k+1], 63.f); if (mx < 0.f) mx += 63.f;
        float sig = (softplus_f(par[2*NK + k] + 2.f) + 0.05f) * (63.f * 0.05f);
        float inv_sig = 1.f / sig;
        int fy = (int)floorf(my), fx = (int)floorf(mx);

        int iy[NVS], ix[NVS];
        iy[0] =  fy      & 63;  ix[0] =  fx      & 63;
        iy[1] =  fy      & 63;  ix[1] = (fx + 1) & 63;
        iy[2] = (fy + 1) & 63;  ix[2] =  fx      & 63;
        iy[3] = (fy + 1) & 63;  ix[3] = (fx + 1) & 63;
        const int* gk = gbase + k * (NGADD*2);
        #pragma unroll
        for (int g = 0; g < NGADD; ++g) { iy[4+g] = gk[2*g]; ix[4+g] = gk[2*g+1]; }
        const int* lk = lbase + k * (NRADD*2);
        #pragma unroll
        for (int r = 0; r < NRADD; ++r) {
            iy[8+r] = (fy + lk[2*r]   - 8) & 63;
            ix[8+r] = (fx + lk[2*r+1] - 8) & 63;
        }

        int   flat[NVS];
        float pr[NVS];
        float s = 0.f;
        #pragma unroll
        for (int v = 0; v < NVS; ++v) {
            flat[v] = (iy[v] << 6) + ix[v];
            float zy = ((float)iy[v] - my) * inv_sig;
            float zx = ((float)ix[v] - mx) * inv_sig;
            float pv = __expf(-0.5f * (zy*zy + zx*zx));
            bool dup = false;
            #pragma unroll
            for (int u = 0; u < v; ++u) dup = dup || (flat[v] == flat[u]);
            pv = dup ? 0.f : pv;
            pr[v] = pv;
            s += pv;
        }
        float invs = 1.f / s;

        if constexpr (FUSED) {
            float feat[NCIN];
            #pragma unroll
            for (int c = 0; c < NCIN; ++c) feat[c] = 0.f;
            #pragma unroll
            for (int v = 0; v < NVS; ++v) {
                float wv = pr[v] * invs;
                const float* xv = xb + flat[v];
                #pragma unroll
                for (int c = 0; c < NCIN; ++c) feat[c] += wv * xv[c << 12];
            }
            for (int c = 0; c < NCIN; ++c) {
                float fv = feat[c];
                const float* wrow = Wu + (size_t)(k*NCIN + c) * NCOUT;
                #pragma unroll
                for (int co = 0; co < NCOUT; ++co) outacc[co] += fv * wrow[co];
            }
        } else {
            // pixel-major gather: 8x b128 per sample point
            v4f feat4[NCIN/4];
            #pragma unroll
            for (int q = 0; q < NCIN/4; ++q) feat4[q] = (v4f)0.f;
            #pragma unroll
            for (int v = 0; v < NVS; ++v) {
                float wv = pr[v] * invs;
                const v4f* xv = (const v4f*)(xT + ((size_t)((b << 12) + flat[v])) * NCIN);
                #pragma unroll
                for (int q = 0; q < NCIN/4; ++q) feat4[q] += wv * xv[q];
            }
            _Float16* dst = reinterpret_cast<_Float16*>(feats_raw)
                          + (size_t)gid * KDIM + k * NCIN;
            #pragma unroll
            for (int q = 0; q < NCIN/4; ++q) {
                v4h hv4 = __builtin_convertvector(feat4[q], v4h);
                *(v4h*)(dst + 4*q) = hv4;
            }
        }
    }

    if constexpr (FUSED) {
        float* ob = out + (((size_t)b * NCOUT) << 12) + p;
        #pragma unroll
        for (int co = 0; co < NCOUT; ++co) ob[(size_t)co << 12] = outacc[co];
    }
}

// ---------------------------------------------------------------------
// Stage 2: out[32768 x 64] = feats[32768 x 256](f16) @ Wu[256 x 64] + bu
// via v_wmma_f32_16x16x32_f16. One 16x16 output tile per wave, 8 WMMAs
// over K=256. Wu transposed to f16 in LDS so each lane's B fragment is a
// contiguous 32B read in this code's k order, same as A (lanes 0-15: K 0-15
// of column n; lanes 16-31: K 16-31).
// ---------------------------------------------------------------------
__global__ __launch_bounds__(256)
void gemm_wmma_kernel(const unsigned short* __restrict__ feats_raw,
                      const float* __restrict__ Wu,
                      const float* __restrict__ bu,
                      float* __restrict__ out)
{
    __shared__ __align__(32) _Float16 sWuT[NCOUT * KDIM];   // [col][k], 32 KB
    const _Float16* feats = reinterpret_cast<const _Float16*>(feats_raw);

    int t = threadIdx.x;
    for (int i = t; i < NCOUT * KDIM; i += 256) {
        int n  = i >> 8;        // column (cout)
        int kk = i & 255;       // k index
        sWuT[i] = (_Float16)Wu[kk * NCOUT + n];
    }
    __syncthreads();

    int wave   = t >> 5;
    int lane   = t & 31;
    int tile   = blockIdx.x * 8 + wave;   // 8192 tiles total
    int tm     = tile >> 2;               // 2048 row tiles
    int tn     = tile & 3;                // 4 col tiles
    int m0     = tm << 4;
    int n0     = tn << 4;
    int lanelo = lane & 15;
    bool hi    = lane >= 16;

    // A fragment source: row m = m0 + (lane&15); lo-lanes K {0-15}(+k0),
    // hi-lanes K {16-31}(+k0) -> two contiguous 16B chunks per step.
    const _Float16* arow = feats + (size_t)(m0 + lanelo) * KDIM + (hi ? 16 : 0);
    // B fragment source: column n = n0 + (lane&15); 16 contiguous K values.
    const _Float16* brow = &sWuT[(n0 + lanelo) * KDIM + (hi ? 16 : 0)];

    float bias = bu[n0 + lanelo];         // bias depends on N only
    v8f acc;
    #pragma unroll
    for (int i = 0; i < 8; ++i) acc[i] = bias;

    __builtin_prefetch(arow, 0, 0);       // global_prefetch_b8

    #pragma unroll
    for (int k0 = 0; k0 < KDIM; k0 += 32) {
        v8h alo = *(const v8h*)(arow + k0);
        v8h ahi = *(const v8h*)(arow + k0 + 8);
        v16h a  = __builtin_shufflevector(alo, ahi,
                    0,1,2,3,4,5,6,7,8,9,10,11,12,13,14,15);
        v16h bf = *(const v16h*)(brow + k0);
        acc = __builtin_amdgcn_wmma_f32_16x16x32_f16(
                  false, a, false, bf, (short)0, acc, false, false);
    }

    // D layout: VGPR v -> M = v (lanes 0-15) / v+8 (lanes 16-31); N = lane&15.
    int bb   = m0 >> 12;                  // batch of this row tile
    int col  = n0 + lanelo;
    float* ob = out + (((size_t)bb * NCOUT + col) << 12);
    int pbase = (m0 & 4095) + (hi ? 8 : 0);
    #pragma unroll
    for (int v = 0; v < 8; ++v) ob[pbase + v] = acc[v];
}

// ---------------------------------------------------------------------
extern "C" void kernel_launch(void* const* d_in, const int* in_sizes, int n_in,
                              void* d_out, int out_size, void* d_ws, size_t ws_size,
                              hipStream_t stream)
{
    const float* x    = (const float*)d_in[0];
    const float* W1   = (const float*)d_in[1];
    const float* b1   = (const float*)d_in[2];
    const float* W2   = (const float*)d_in[3];
    const float* b2   = (const float*)d_in[4];
    const float* Wu   = (const float*)d_in[5];
    const float* bu   = (const float*)d_in[6];
    const int*   gidx = (const int*)d_in[7];
    const int*   loff = (const int*)d_in[8];
    float* out = (float*)d_out;

    const size_t featB = (size_t)NPIX * KDIM * 2;        // 16 MB f16 feats
    const size_t xtB   = (size_t)NPIX * NCIN * 4;        //  4 MB f32 xT
    if (ws_size >= featB + xtB) {
        unsigned short* feats = (unsigned short*)d_ws;
        float* xT = (float*)((char*)d_ws + featB);
        transpose_x_kernel<<<NPIX/256, 256, 0, stream>>>(x, xT);
        stage1_kernel<false><<<NPIX/256, 256, 0, stream>>>(
            x, xT, W1, b1, W2, b2, gidx, loff, feats, Wu, bu, nullptr);
        // 8192 wave-tiles / 8 waves per block = 1024 blocks
        gemm_wmma_kernel<<<1024, 256, 0, stream>>>(feats, Wu, bu, out);
    } else {
        stage1_kernel<true><<<NPIX/256, 256, 0, stream>>>(
            x, nullptr, W1, b1, W2, b2, gidx, loff, nullptr, Wu, bu, out);
    }
}
